// MambaBlock_76424648065115
// MI455X (gfx1250) — matrix operand order repaired and checked by reading, not compiled, not run on hardware
//
#include <hip/hip_runtime.h>
#include <hip/hip_bf16.h>
#include <math.h>

#define D_STATE 16
#define D_CONV 4
#define DT_RANK 64
#define D_INNER 2048
#define CHANNELS 1024
#define SEQ_T 1024
#define BATCH 2

typedef __attribute__((ext_vector_type(16))) __bf16 v16bf;
typedef __attribute__((ext_vector_type(8)))  __bf16 v8bf;
typedef __attribute__((ext_vector_type(8)))  float  v8f;

// ---------------------------------------------------------------------------
// f32 -> bf16 elementwise (used once per launch for the 4 weight matrices)
// ---------------------------------------------------------------------------
__global__ __launch_bounds__(256) void cvt_bf16(
    const float* __restrict__ in, __bf16* __restrict__ out, int n)
{
    int i = blockIdx.x * blockDim.x + threadIdx.x;
    if (i < n) out[i] = (__bf16)in[i];
}

// ---------------------------------------------------------------------------
// WMMA fragment loaders (layouts per CDNA5 ISA §7.12.2, wave32)
// Operands are bf16 in memory -> pure vector loads, no converts in hot loop.
// A: 16x32 (MxK). lane L: M = L%16; elems 0..7 -> K = kb+0..7,
//    elems 8..15 -> K = 16+kb+0..7, kb = (L/16)*8.  (2 x 16B loads)
// ---------------------------------------------------------------------------
__device__ inline v16bf load_frag_a(const __bf16* __restrict__ A, int lda,
                                    int m0, int k0, int lane) {
    int m  = m0 + (lane & 15);
    int kb = (lane >> 4) * 8;
    const __bf16* p = A + (size_t)m * lda + k0 + kb;
    v8bf lo = *(const v8bf*)p;
    v8bf hi = *(const v8bf*)(p + 16);
    v16bf a;
#pragma unroll
    for (int j = 0; j < 8; ++j) { a[j] = lo[j]; a[8 + j] = hi[j]; }
    return a;
}

// B: 32x16 (KxN) from row-major W (N x K): B[k][n] = W[n][k].
// lane L: N = L%16; elems 0..7 -> K = kb+0..7, elems 8..15 -> K = 16+kb+0..7, kb = (L/16)*8.  (2 x 16B loads)
__device__ inline v16bf load_frag_b(const __bf16* __restrict__ W, int ldb,
                                    int n0, int k0, int lane) {
    int n  = n0 + (lane & 15);
    int kb = (lane >> 4) * 8;
    const __bf16* p = W + (size_t)n * ldb + k0 + kb;
    v8bf lo = *(const v8bf*)p;
    v8bf hi = *(const v8bf*)(p + 16);
    v16bf b;
#pragma unroll
    for (int j = 0; j < 8; ++j) { b[j] = lo[j]; b[8 + j] = hi[j]; }
    return b;
}

// ---------------------------------------------------------------------------
// Generic WMMA GEMM: Out[M x N] = A[M x K] * W[N x K]^T  (both K-contiguous)
// EPI 0: f32 store.  EPI 1: softplus(acc + bias[n]) f32 store.
// EPI 2: f32 store + bf16 dual-store of columns n < ldob (dt_raw slice).
// Block = 256 thr = 8 waves (4 in M, 2 in N). Block tile 64(M) x 128(N).
// Wave tile 16 x 64 (A fragment reused across 4 WMMAs).
// ---------------------------------------------------------------------------
template <int EPI>
__global__ __launch_bounds__(256) void gemm_wmma(
    const __bf16* __restrict__ A, const __bf16* __restrict__ W,
    const float* __restrict__ bias, float* __restrict__ Out,
    __bf16* __restrict__ OutB,
    int M, int N, int K, int lda, int ldb, int ldo, int ldob)
{
    int lane = threadIdx.x & 31;
    int wave = threadIdx.x >> 5;
    int wm = wave & 3, wn = wave >> 2;
    int m0 = blockIdx.y * 64 + wm * 16;
    int n0 = blockIdx.x * 128 + wn * 64;

    v8f acc[4];
#pragma unroll
    for (int i = 0; i < 4; ++i)
#pragma unroll
        for (int r = 0; r < 8; ++r) acc[i][r] = 0.0f;

    for (int k0 = 0; k0 < K; k0 += 32) {
        v16bf afrag = load_frag_a(A, lda, m0, k0, lane);
#pragma unroll
        for (int i = 0; i < 4; ++i) {
            int nsub = n0 + 16 * i;
            int nc = (nsub < N) ? nsub : 0;     // wave-uniform clamp, EXEC stays full
            v16bf bfrag = load_frag_b(W, ldb, nc, k0, lane);
            acc[i] = __builtin_amdgcn_wmma_f32_16x16x32_bf16(
                false, afrag, false, bfrag, (short)0, acc[i], false, false);
        }
    }

    // D layout: lane L -> N = L%16, M = r + 8*(L/16)
    int nlane = lane & 15;
    int mrow  = m0 + (lane >> 4) * 8;
#pragma unroll
    for (int i = 0; i < 4; ++i) {
        int n = n0 + 16 * i + nlane;
        if (n >= N) continue;
#pragma unroll
        for (int r = 0; r < 8; ++r) {
            float v = acc[i][r];
            if (EPI == 1) {
                v += bias[n];
                v = (v > 20.0f) ? v : log1pf(__expf(v));  // softplus
            }
            size_t m = (size_t)(mrow + r);
            Out[m * ldo + n] = v;
            if (EPI == 2 && n < ldob) OutB[m * ldob + n] = (__bf16)v;
        }
    }
}

// ---------------------------------------------------------------------------
// LayerNorm pass 1: per (b,t) mean / rstd over C. Thread per t -> coalesced.
// ---------------------------------------------------------------------------
__global__ __launch_bounds__(256) void ln_stats(
    const float* __restrict__ x, float* __restrict__ mu,
    float* __restrict__ rstd)
{
    int idx = blockIdx.x * blockDim.x + threadIdx.x;      // b*T + t
    int b = idx >> 10, t = idx & (SEQ_T - 1);
    const float* p = x + (size_t)b * CHANNELS * SEQ_T + t;
    float s = 0.0f, s2 = 0.0f;
    for (int c = 0; c < CHANNELS; ++c) {
        float v = p[(size_t)c * SEQ_T];
        s += v; s2 += v * v;
    }
    float m = s * (1.0f / CHANNELS);
    float var = s2 * (1.0f / CHANNELS) - m * m;
    mu[idx] = m;
    rstd[idx] = rsqrtf(var + 1e-5f);
}

// LN pass 2 + transpose (B,C,T) -> (B*T, C), bf16 out for GEMM1 A operand.
__global__ __launch_bounds__(256) void ln_apply(
    const float* __restrict__ x, const float* __restrict__ mu,
    const float* __restrict__ rstd, const float* __restrict__ w,
    const float* __restrict__ bias, __bf16* __restrict__ xn)
{
    __shared__ float tile[32][33];
    int b  = blockIdx.z;
    int c0 = blockIdx.y * 32;
    int t0 = blockIdx.x * 32;
    int tx = threadIdx.x & 31;
    int ty = threadIdx.x >> 5;                 // 0..7
    const float* xb = x + (size_t)b * CHANNELS * SEQ_T;
#pragma unroll
    for (int i = 0; i < 4; ++i) {              // read coalesced along t
        int c = c0 + ty + i * 8;
        tile[ty + i * 8][tx] = xb[(size_t)c * SEQ_T + t0 + tx];
    }
    __syncthreads();
#pragma unroll
    for (int i = 0; i < 4; ++i) {              // write coalesced along c
        int t = t0 + ty + i * 8;
        int c = c0 + tx;
        int row = b * SEQ_T + t;
        float v = (tile[tx][ty + i * 8] - mu[row]) * rstd[row] * w[c] + bias[c];
        xn[(size_t)row * CHANNELS + c] = (__bf16)v;
    }
}

// ---------------------------------------------------------------------------
// Depthwise causal conv (width 4) + bias + SiLU. x_in = xz[:, 0:2048].
// Dual store: f32 for the scan, bf16 for the x_proj GEMM A operand.
// ---------------------------------------------------------------------------
__global__ __launch_bounds__(256) void conv_silu(
    const float* __restrict__ xz, const float* __restrict__ cw,
    const float* __restrict__ cb, float* __restrict__ xc,
    __bf16* __restrict__ xcb)
{
    int idx = blockIdx.x * blockDim.x + threadIdx.x;   // over B*T*D_INNER
    int d   = idx & (D_INNER - 1);
    int row = idx >> 11;                               // b*T + t
    int t   = row & (SEQ_T - 1);
    float acc = cb[d];
#pragma unroll
    for (int j = 0; j < D_CONV; ++j) {
        int tt = t - (D_CONV - 1) + j;
        if (tt >= 0)
            acc += cw[d * D_CONV + j] *
                   xz[(size_t)(row - (D_CONV - 1) + j) * (2 * D_INNER) + d];
    }
    float s = acc / (1.0f + __expf(-acc));             // SiLU
    xc[idx]  = s;
    xcb[idx] = (__bf16)s;
}

// ---------------------------------------------------------------------------
// SSM selective scan. Lane per (b,d,s): 65536 lanes = 2048 waves of latency
// hiding for the T-sequential recurrence. 16-lane xor-butterfly reduces the
// state dot product each step. Fused D*x_c skip + silu(z) gate; bf16 output
// feeds the out_proj GEMM directly.
// ---------------------------------------------------------------------------
__global__ __launch_bounds__(256) void ssm_scan(
    const float* __restrict__ dtb, const float* __restrict__ xc,
    const float* __restrict__ xdbl, const float* __restrict__ xz,
    const float* __restrict__ A_log, const float* __restrict__ Dp,
    __bf16* __restrict__ y)
{
    int lane = threadIdx.x & 31;
    int s    = lane & 15;
    int half = lane >> 4;
    int wave = threadIdx.x >> 5;
    int ch = blockIdx.x * 16 + wave * 2 + half;        // 0 .. B*D_INNER-1
    int b = ch >> 11;
    int d = ch & (D_INNER - 1);

    float Aval = -__expf(A_log[d * D_STATE + s]);
    float Dval = Dp[d];
    float h = 0.0f;
    int rowbase = b * SEQ_T;

    for (int t = 0; t < SEQ_T; ++t) {
        int row = rowbase + t;
        float dtv = dtb[(size_t)row * D_INNER + d];    // broadcast within 16 lanes
        float xcv = xc[(size_t)row * D_INNER + d];
        float Bv  = xdbl[row * 96 + DT_RANK + s];      // coalesced 16 lanes
        float Cv  = xdbl[row * 96 + DT_RANK + D_STATE + s];
        float dA  = __expf(dtv * Aval);
        h = fmaf(dA, h, dtv * Bv * xcv);
        float p = h * Cv;
        p += __shfl_xor(p, 1, 16);
        p += __shfl_xor(p, 2, 16);
        p += __shfl_xor(p, 4, 16);
        p += __shfl_xor(p, 8, 16);
        if (s == 0) {
            float zv = xz[(size_t)row * (2 * D_INNER) + D_INNER + d];
            float sz = zv / (1.0f + __expf(-zv));      // silu(z)
            y[(size_t)row * D_INNER + d] = (__bf16)((p + Dval * xcv) * sz);
        }
    }
}

// ---------------------------------------------------------------------------
// Epilogue: (B*T, C) -> (B, C, T) transpose + residual, LDS tiled.
// ---------------------------------------------------------------------------
__global__ __launch_bounds__(256) void out_transpose_residual(
    const float* __restrict__ ot, const float* __restrict__ x,
    float* __restrict__ out)
{
    __shared__ float tile[32][33];
    int b  = blockIdx.z;
    int c0 = blockIdx.y * 32;
    int t0 = blockIdx.x * 32;
    int tx = threadIdx.x & 31;
    int ty = threadIdx.x >> 5;
#pragma unroll
    for (int i = 0; i < 4; ++i) {              // read coalesced along c
        int t = t0 + ty + i * 8;
        tile[ty + i * 8][tx] = ot[(size_t)(b * SEQ_T + t) * CHANNELS + c0 + tx];
    }
    __syncthreads();
#pragma unroll
    for (int i = 0; i < 4; ++i) {              // write coalesced along t
        int c = c0 + ty + i * 8;
        size_t off = (size_t)b * CHANNELS * SEQ_T + (size_t)c * SEQ_T + t0 + tx;
        out[off] = tile[tx][ty + i * 8] + x[off];
    }
}

// ---------------------------------------------------------------------------
extern "C" void kernel_launch(void* const* d_in, const int* in_sizes, int n_in,
                              void* d_out, int out_size, void* d_ws, size_t ws_size,
                              hipStream_t stream)
{
    const float* x         = (const float*)d_in[0];
    const float* norm_w    = (const float*)d_in[1];
    const float* norm_b    = (const float*)d_in[2];
    const float* in_proj_w = (const float*)d_in[3];   // (4096, 1024)
    const float* conv_w    = (const float*)d_in[4];   // (2048, 1, 4)
    const float* conv_b    = (const float*)d_in[5];
    const float* x_proj_w  = (const float*)d_in[6];   // (96, 2048)
    const float* dt_proj_w = (const float*)d_in[7];   // (2048, 64)
    const float* dt_proj_b = (const float*)d_in[8];
    const float* A_log     = (const float*)d_in[9];   // (2048, 16)
    const float* Dp        = (const float*)d_in[10];
    const float* out_pw    = (const float*)d_in[11];  // (1024, 2048)
    float* out = (float*)d_out;

    const int BT = BATCH * SEQ_T;                     // 2048 rows
    float* ws = (float*)d_ws;
    size_t o = 0;
    float* mu   = ws + o; o += BT;
    float* rstd = ws + o; o += BT;
    float* xz   = ws + o; o += (size_t)BT * 2 * D_INNER;     // 32 MB
    float* xc   = ws + o; o += (size_t)BT * D_INNER;         // 16 MB
    float* xdbl = ws + o; o += (size_t)BT * 96;
    float* dtb  = ws + o; o += (size_t)BT * D_INNER;         // 16 MB
    float* otmp = ws + o; o += (size_t)BT * CHANNELS;        //  8 MB
    // bf16 regions (sizes counted in floats = elems/2; all element counts even)
    __bf16* xn    = (__bf16*)(ws + o); o += (size_t)BT * CHANNELS / 2;
    __bf16* xcb   = (__bf16*)(ws + o); o += (size_t)BT * D_INNER / 2;
    __bf16* dtraw = (__bf16*)(ws + o); o += (size_t)BT * DT_RANK / 2;
    __bf16* ybuf  = (__bf16*)(ws + o); o += (size_t)BT * D_INNER / 2;
    __bf16* w_in  = (__bf16*)(ws + o); o += (size_t)2 * D_INNER * CHANNELS / 2;
    __bf16* w_x   = (__bf16*)(ws + o); o += (size_t)96 * D_INNER / 2;
    __bf16* w_dt  = (__bf16*)(ws + o); o += (size_t)D_INNER * DT_RANK / 2;
    __bf16* w_out = (__bf16*)(ws + o); o += (size_t)CHANNELS * D_INNER / 2;

    // 0) weights -> bf16 (once per launch; deterministic)
    int n_in_w  = 2 * D_INNER * CHANNELS;
    int n_x_w   = 96 * D_INNER;
    int n_dt_w  = D_INNER * DT_RANK;
    int n_out_w = CHANNELS * D_INNER;
    cvt_bf16<<<(n_in_w  + 255) / 256, 256, 0, stream>>>(in_proj_w, w_in,  n_in_w);
    cvt_bf16<<<(n_x_w   + 255) / 256, 256, 0, stream>>>(x_proj_w,  w_x,   n_x_w);
    cvt_bf16<<<(n_dt_w  + 255) / 256, 256, 0, stream>>>(dt_proj_w, w_dt,  n_dt_w);
    cvt_bf16<<<(n_out_w + 255) / 256, 256, 0, stream>>>(out_pw,    w_out, n_out_w);

    // 1) LayerNorm stats + apply/transpose -> xn bf16 (BT x 1024)
    ln_stats<<<BT / 256, 256, 0, stream>>>(x, mu, rstd);
    ln_apply<<<dim3(SEQ_T / 32, CHANNELS / 32, BATCH), 256, 0, stream>>>(
        x, mu, rstd, norm_w, norm_b, xn);

    // 2) xz = xn @ in_proj_w^T   (2048 x 4096, K=1024)
    gemm_wmma<0><<<dim3(4096 / 128, 2048 / 64), 256, 0, stream>>>(
        xn, w_in, nullptr, xz, nullptr, BT, 2 * D_INNER, CHANNELS,
        CHANNELS, CHANNELS, 2 * D_INNER, 0);

    // 3) depthwise conv + SiLU -> xc f32 + xcb bf16
    conv_silu<<<(BT * D_INNER) / 256, 256, 0, stream>>>(xz, conv_w, conv_b, xc, xcb);

    // 4) x_dbl = xc @ x_proj_w^T (2048 x 96, K=2048); dual-store dt_raw bf16
    gemm_wmma<2><<<dim3(1, 2048 / 64), 256, 0, stream>>>(
        xcb, w_x, nullptr, xdbl, dtraw, BT, 96, D_INNER,
        D_INNER, D_INNER, 96, DT_RANK);

    // 5) dt = softplus(dt_raw @ dt_proj_w^T + b)   (2048 x 2048, K=64)
    gemm_wmma<1><<<dim3(2048 / 128, 2048 / 64), 256, 0, stream>>>(
        dtraw, w_dt, dt_proj_b, dtb, nullptr, BT, D_INNER, DT_RANK,
        DT_RANK, DT_RANK, D_INNER, 0);

    // 6) selective scan + skip + silu(z) gate -> ybuf bf16 (BT x 2048)
    ssm_scan<<<(BATCH * D_INNER) / 16, 256, 0, stream>>>(
        dtb, xc, xdbl, xz, A_log, Dp, ybuf);

    // 7) otmp = ybuf @ out_proj_w^T   (2048 x 1024, K=2048)
    gemm_wmma<0><<<dim3(1024 / 128, 2048 / 64), 256, 0, stream>>>(
        ybuf, w_out, nullptr, otmp, nullptr, BT, CHANNELS, D_INNER,
        D_INNER, D_INNER, CHANNELS, 0);

    // 8) transpose back to (B, C, T) + residual
    out_transpose_residual<<<dim3(SEQ_T / 32, CHANNELS / 32, BATCH), 256, 0, stream>>>(
        otmp, x, out);
}
